// MoE_82454782148671
// MI455X (gfx1250) — matrix operand order repaired and checked by reading, not compiled, not run on hardware
//
#include <hip/hip_runtime.h>
#include <cstdint>
#include <cstddef>

#define DIM      2048
#define HID      1408
#define NE       16
#define TOPK     4
#define SHID     2816
#define T_TOK    2048

typedef __attribute__((ext_vector_type(16))) __bf16 v16bf;
typedef __attribute__((ext_vector_type(8)))  float  v8f;
typedef unsigned int v4u __attribute__((ext_vector_type(4)));
typedef int          v8i __attribute__((ext_vector_type(8)));
typedef int          v4i __attribute__((ext_vector_type(4)));

#ifndef __has_builtin
#define __has_builtin(x) 0
#endif
#if __has_builtin(__builtin_amdgcn_tensor_load_to_lds) && \
    __has_builtin(__builtin_amdgcn_s_wait_tensorcnt)
#define USE_TDM 1
#else
#define USE_TDM 0
#endif

// ---- f32 -> bf16 with round-to-nearest-even (pure integer ops, always lowers) ----
__device__ __forceinline__ unsigned short f2bf(float f) {
    unsigned int u = __float_as_uint(f);
    u += 0x7FFFu + ((u >> 16) & 1u);
    return (unsigned short)(u >> 16);
}

union FragU { v16bf v; uint4 u[2]; };

// A-matrix fragment (16x32 bf16, row-major tile of 32 cols in LDS).
// ISA 7.12.2: lane L holds row (L&15); K = hi*8 + {0..7} in v[0..3], hi*8+16+{0..7} in v[4..7].
__device__ __forceinline__ v16bf load_frag_a(const unsigned short* tile, int row0, int lane) {
    int r  = lane & 15;
    int kb = (lane >> 4) << 3;          // 0 or 8
    const unsigned short* p = tile + (row0 + r) * 32 + kb;
    FragU f;
    f.u[0] = *(const uint4*)(p);
    f.u[1] = *(const uint4*)(p + 16);
    return f.v;
}

// B-matrix fragment (32x16 bf16). We stage W row-major as [n][k] so B[k][n] = Ws[n][k].
// ISA: one B row striped across lanes within one VGPR -> lane L holds col (L&15),
// K = hi*8 + {0..7} in v[0..3], hi*8+16+{0..7} in v[4..7] (same as A).
__device__ __forceinline__ v16bf load_frag_b(const unsigned short* tile, int row0, int lane) {
    int r  = lane & 15;
    int kb = (lane >> 4) << 3;          // 0 or 8
    const unsigned short* p = tile + (row0 + r) * 32 + kb;
    FragU f;
    f.u[0] = *(const uint4*)(p);
    f.u[1] = *(const uint4*)(p + 16);
    return f.v;
}

// ---------------------------------------------------------------------------
// Gate: per-token logits -> softmax -> top-4. One block (256 thr) per token.
// ---------------------------------------------------------------------------
__global__ void __launch_bounds__(256)
gate_kernel(const float* __restrict__ x, const float* __restrict__ gw,
            int* __restrict__ idx, float* __restrict__ wts,
            float* __restrict__ scores) {
    __shared__ float red[NE][16];
    __shared__ float lg[NE];
    const int t = blockIdx.x;
    const int tid = threadIdx.x;
    const int e = tid >> 4, c = tid & 15;
    const float* xt = x + (size_t)t * DIM;
    const float* ge = gw + (size_t)e * DIM;
    float p = 0.f;
    const int j0 = c * (DIM / 16);
    #pragma unroll 8
    for (int j = 0; j < DIM / 16; ++j) p += xt[j0 + j] * ge[j0 + j];
    red[e][c] = p;
    __syncthreads();
    if (c == 0) {
        float s = 0.f;
        for (int i = 0; i < 16; ++i) s += red[e][i];
        lg[e] = s;
    }
    __syncthreads();
    if (tid == 0) {
        float mx = lg[0];
        for (int i = 1; i < NE; ++i) mx = fmaxf(mx, lg[i]);
        float sm[NE]; float den = 0.f;
        for (int i = 0; i < NE; ++i) { sm[i] = __expf(lg[i] - mx); den += sm[i]; }
        const float inv = 1.f / den;
        float tmp[NE];
        for (int i = 0; i < NE; ++i) {
            sm[i] *= inv;
            scores[(size_t)t * NE + i] = sm[i];
            tmp[i] = sm[i];
        }
        for (int k = 0; k < TOPK; ++k) {          // top-4, first-index tie break
            int bi = 0; float bv = tmp[0];
            for (int i = 1; i < NE; ++i) if (tmp[i] > bv) { bv = tmp[i]; bi = i; }
            idx[t * TOPK + k] = bi;
            wts[t * TOPK + k] = bv;               // ROUTE_SCALE == 1.0
            tmp[bi] = -1.f;
        }
    }
}

// ---------------------------------------------------------------------------
// Build stable per-expert token lists (+weights), prefix offsets, identity
// list for the shared MLP. One wave; thread e scans tokens in order.
// ---------------------------------------------------------------------------
__global__ void build_lists_kernel(const int* __restrict__ idx, const float* __restrict__ wts,
                                   int* __restrict__ lists, float* __restrict__ wlists,
                                   int* __restrict__ counts, int* __restrict__ offsets,
                                   int* __restrict__ ident, int* __restrict__ cntShared,
                                   int* __restrict__ offShared) {
    const int tid = threadIdx.x;  // 32
    if (tid < NE) {
        const int e = tid;
        int c = 0;
        for (int t = 0; t < T_TOK; ++t)
            for (int k = 0; k < TOPK; ++k)
                if (idx[t * TOPK + k] == e) {
                    lists[e * T_TOK + c]  = t;
                    wlists[e * T_TOK + c] = wts[t * TOPK + k];
                    ++c;
                }
        counts[e] = c;
    }
    for (int t = tid; t < T_TOK; t += 32) ident[t] = t;
    __syncthreads();
    if (tid == 0) {
        int acc = 0;
        for (int e = 0; e < NE; ++e) { offsets[e] = acc; acc += counts[e]; }
        offsets[NE] = acc;                        // == T_TOK*TOPK
        cntShared[0] = T_TOK;
        offShared[0] = 0;
    }
}

// ---------------------------------------------------------------------------
// Aux load-balance loss (deterministic fixed-tree reduction).
// ---------------------------------------------------------------------------
__global__ void __launch_bounds__(256)
aux_kernel(const float* __restrict__ scores, const int* __restrict__ counts,
           float* __restrict__ out_aux) {
    __shared__ float red[NE][16];
    const int tid = threadIdx.x;
    const int e = tid >> 4, c = tid & 15;
    float s = 0.f;
    for (int t = c; t < T_TOK; t += 16) s += scores[(size_t)t * NE + e];
    red[e][c] = s;
    __syncthreads();
    if (tid == 0) {
        float aux = 0.f;
        for (int i = 0; i < NE; ++i) {
            float se = 0.f;
            for (int j = 0; j < 16; ++j) se += red[i][j];
            const float tpe = (float)counts[i] / (float)(T_TOK * TOPK);
            aux += tpe * (se / (float)T_TOK);
        }
        out_aux[0] = aux * (float)NE;
    }
}

// ---------------------------------------------------------------------------
// Pass 1: h = silu(x W1^T) * (x W3^T) * route_weight  for gathered tokens.
// Block = 128 tokens x 64 hidden, 8 waves, K staged 32-wide through LDS.
// Emits v_wmma_f32_16x16x32_bf16 (2 GEMMs fused). Output packed bf16.
// ---------------------------------------------------------------------------
__global__ void __launch_bounds__(256)
moe_h_kernel(const float* __restrict__ x,
             const float* __restrict__ W1, const float* __restrict__ W3,
             const int* __restrict__ lists, const float* __restrict__ wlists,
             const int* __restrict__ counts, const int* __restrict__ offsets,
             unsigned short* __restrict__ Hbuf, int hid, long long wstride) {
    __shared__ __align__(16) unsigned short Xs[128 * 32];
    __shared__ __align__(16) unsigned short W1s[64 * 32];
    __shared__ __align__(16) unsigned short W3s[64 * 32];
    __shared__ int   tok[128];
    __shared__ float wgt[128];

    const int e   = blockIdx.z;
    const int cnt = counts[e];
    const int m0  = blockIdx.x * 128;
    if (m0 >= cnt) return;
    const int n0    = blockIdx.y * 64;
    const int hbase = offsets[e];
    const int tid   = threadIdx.x;
    const float* W1e = W1 + (size_t)e * wstride;
    const float* W3e = W3 + (size_t)e * wstride;
    const int*   list = lists + (size_t)e * T_TOK;
    const float* wl   = wlists ? (wlists + (size_t)e * T_TOK) : nullptr;

    if (tid < 128) {
        const int r  = tid;
        const int tk = (m0 + r < cnt) ? list[m0 + r] : -1;
        tok[r] = tk;
        wgt[r] = (tk >= 0) ? (wl ? wl[m0 + r] : 1.0f) : 0.0f;
    }
    __syncthreads();

    const int wave = tid >> 5, lane = tid & 31;
    v8f a1[4] = {};
    v8f a3[4] = {};

    for (int k0 = 0; k0 < DIM; k0 += 32) {
        {   // stage X tile: 128x32, 2 threads/row
            const int r = tid >> 1, cb = (tid & 1) * 16;
            const int tk = tok[r];
            if (tk >= 0) {
                const float* src = x + (size_t)tk * DIM + k0 + cb;
                #pragma unroll
                for (int j = 0; j < 16; ++j) Xs[r * 32 + cb + j] = f2bf(src[j]);
            } else {
                #pragma unroll
                for (int j = 0; j < 16; ++j) Xs[r * 32 + cb + j] = 0;
            }
        }
        {   // stage W1/W3 tiles: 64x32, 4 threads/row
            const int r = tid >> 2, cb = (tid & 3) * 8;
            const float* s1 = W1e + (size_t)(n0 + r) * DIM + k0 + cb;
            const float* s3 = W3e + (size_t)(n0 + r) * DIM + k0 + cb;
            if (k0 + 32 < DIM) {                  // pull next K-slab toward L2
                __builtin_prefetch(s1 + 32, 0, 0);
                __builtin_prefetch(s3 + 32, 0, 0);
            }
            #pragma unroll
            for (int j = 0; j < 8; ++j) {
                W1s[r * 32 + cb + j] = f2bf(s1[j]);
                W3s[r * 32 + cb + j] = f2bf(s3[j]);
            }
        }
        __syncthreads();
        const v16bf a = load_frag_a(Xs, wave * 16, lane);
        #pragma unroll
        for (int s = 0; s < 4; ++s) {
            const v16bf b1 = load_frag_b(W1s, s * 16, lane);
            const v16bf b3 = load_frag_b(W3s, s * 16, lane);
            a1[s] = __builtin_amdgcn_wmma_f32_16x16x32_bf16(false, a, false, b1,
                                                            (short)0, a1[s], false, false);
            a3[s] = __builtin_amdgcn_wmma_f32_16x16x32_bf16(false, a, false, b3,
                                                            (short)0, a3[s], false, false);
        }
        __syncthreads();
    }

    // epilogue: SwiGLU + route-weight scale, store packed bf16
    const int col = lane & 15, hi = lane >> 4;
    #pragma unroll
    for (int s = 0; s < 4; ++s) {
        #pragma unroll
        for (int r = 0; r < 8; ++r) {
            const int lrow = wave * 16 + hi * 8 + r;
            if (m0 + lrow < cnt) {
                const float g = a1[s][r];
                const float h = (g / (1.f + __expf(-g))) * a3[s][r] * wgt[lrow];
                Hbuf[(size_t)(hbase + m0 + lrow) * hid + n0 + s * 16 + col] = f2bf(h);
            }
        }
    }
}

// ---------------------------------------------------------------------------
// Pass 2: y[t,:] += h W2^T   (h already bf16 + weight-scaled).
// Block = 128 tokens x 64 output dims; atomicAdd combine into d_out.
// The A-tile (packed bf16, plain 2D strided block) is DMA'd by the Tensor
// Data Mover: one wave issues TENSOR_LOAD_TO_LDS per K-step; TDM OOB rules
// (tensor_dim1 = remaining rows) zero-fill the ragged tail for free.
// ---------------------------------------------------------------------------
__global__ void __launch_bounds__(256)
moe_out_kernel(const unsigned short* __restrict__ Hbuf, const float* __restrict__ W2,
               const int* __restrict__ lists, const int* __restrict__ counts,
               const int* __restrict__ offsets, float* __restrict__ out,
               int hid, long long w2stride) {
    __shared__ __align__(16) unsigned short As[128 * 32];
    __shared__ __align__(16) unsigned short Ws[64 * 32];
    __shared__ int tok[128];

    const int e   = blockIdx.z;
    const int cnt = counts[e];
    const int m0  = blockIdx.x * 128;
    if (m0 >= cnt) return;
    const int n0    = blockIdx.y * 64;
    const int hbase = offsets[e];
    const int tid   = threadIdx.x;
    const float* W2e  = W2 + (size_t)e * w2stride;
    const int*   list = lists + (size_t)e * T_TOK;

    if (tid < 128) tok[tid] = (m0 + tid < cnt) ? list[m0 + tid] : -1;
    __syncthreads();

    const int wave = tid >> 5, lane = tid & 31;
    v8f acc[4] = {};

#if USE_TDM
    const unsigned lds_base = (unsigned)(uintptr_t)(void*)As;   // LDS byte address
    const unsigned long long gbase =
        (unsigned long long)(uintptr_t)(Hbuf + (size_t)(hbase + m0) * hid);
    const unsigned td1 = (unsigned)(cnt - m0);                  // remaining rows
#endif

    for (int k0 = 0; k0 < hid; k0 += 32) {
#if USE_TDM
        if (wave == 0) {
            // ---- D# group 0: count=1, lds_addr, 57-bit global addr, type=2 ----
            const unsigned long long ga = gbase + (unsigned long long)k0 * 2ull;
            v4u g0;
            g0.x = 1u;                                            // count=1, user mode
            g0.y = lds_base;                                      // lds_addr
            g0.z = (unsigned)(ga & 0xFFFFFFFFu);                  // global_addr[31:0]
            g0.w = (unsigned)((ga >> 32) & 0x01FFFFFFu) | (2u << 30); // addr[56:32]|type=2
            // ---- D# group 1: data_size=2B, 2D tile 32x128, dim0 stride = hid ----
            const unsigned td0 = (unsigned)(hid - k0);            // remaining cols
            v8i g1;
            g1[0] = (int)(1u << 16);                              // data_size=1 (2 bytes)
            g1[1] = (int)((td0 & 0xFFFFu) << 16);                 // tensor_dim0[15:0]
            g1[2] = (int)(((td0 >> 16) & 0xFFFFu) | ((td1 & 0xFFFFu) << 16));
            g1[3] = (int)(((td1 >> 16) & 0xFFFFu) | (32u << 16)); // tile_dim0 = 32
            g1[4] = (int)128u;                                    // tile_dim1=128, dim2=0
            g1[5] = (int)(unsigned)hid;                           // dim0_stride[31:0]
            g1[6] = 0;                                            // stride hi / dim1_stride
            g1[7] = 0;
            v4i g2 = {0, 0, 0, 0};
            v4i g3 = {0, 0, 0, 0};
            v8i g4 = {0, 0, 0, 0, 0, 0, 0, 0};                    // unused (VADDR4 slot)
            __builtin_amdgcn_tensor_load_to_lds(g0, g1, g2, g3, g4, 0);
        }
#else
        {   // manual A stage (already bf16) with 16B vector copies
            const int r = tid >> 1, cb = (tid & 1) * 16;
            uint4* d4 = (uint4*)(As + r * 32 + cb);
            if (m0 + r < cnt) {
                const uint4* s4 =
                    (const uint4*)(Hbuf + (size_t)(hbase + m0 + r) * hid + k0 + cb);
                d4[0] = s4[0];
                d4[1] = s4[1];
            } else {
                uint4 z; z.x = z.y = z.z = z.w = 0u;
                d4[0] = z; d4[1] = z;
            }
        }
#endif
        {   // stage W2 tile: 64x32 (overlaps with TDM DMA of the A tile)
            const int r = tid >> 2, cb = (tid & 3) * 8;
            const float* s2 = W2e + (size_t)(n0 + r) * hid + k0 + cb;
            if (k0 + 32 < hid) __builtin_prefetch(s2 + 32, 0, 0);
            #pragma unroll
            for (int j = 0; j < 8; ++j) Ws[r * 32 + cb + j] = f2bf(s2[j]);
        }
#if USE_TDM
        if (wave == 0) __builtin_amdgcn_s_wait_tensorcnt(0);
#endif
        __syncthreads();
        const v16bf a = load_frag_a(As, wave * 16, lane);
        #pragma unroll
        for (int s = 0; s < 4; ++s) {
            const v16bf b = load_frag_b(Ws, s * 16, lane);
            acc[s] = __builtin_amdgcn_wmma_f32_16x16x32_bf16(false, a, false, b,
                                                             (short)0, acc[s], false, false);
        }
        __syncthreads();
    }

    const int col = lane & 15, hi = lane >> 4;
    #pragma unroll
    for (int s = 0; s < 4; ++s) {
        #pragma unroll
        for (int r = 0; r < 8; ++r) {
            const int lrow = wave * 16 + hi * 8 + r;
            const int tk = tok[lrow];
            if (tk >= 0)
                atomicAdd(&out[(size_t)tk * DIM + n0 + s * 16 + col], acc[s][r]);
        }
    }
}

__global__ void zero_kernel(float* __restrict__ p, int n) {
    const int i = blockIdx.x * blockDim.x + threadIdx.x;
    if (i < n) p[i] = 0.f;
}

// ---------------------------------------------------------------------------
extern "C" void kernel_launch(void* const* d_in, const int* in_sizes, int n_in,
                              void* d_out, int out_size, void* d_ws, size_t ws_size,
                              hipStream_t stream) {
    (void)in_sizes; (void)n_in; (void)ws_size;
    const float* x   = (const float*)d_in[0];
    const float* gw  = (const float*)d_in[1];
    const float* w1  = (const float*)d_in[2];
    const float* w2  = (const float*)d_in[3];
    const float* w3  = (const float*)d_in[4];
    const float* sw1 = (const float*)d_in[5];
    const float* sw2 = (const float*)d_in[6];
    const float* sw3 = (const float*)d_in[7];
    float* out = (float*)d_out;

    // workspace carve-up (256B aligned blocks)
    char* ws = (char*)d_ws;
    size_t off = 0;
    auto carve = [&](size_t bytes) -> void* {
        void* p = ws + off;
        off = (off + bytes + 255) & ~(size_t)255;
        return p;
    };
    int*            idx       = (int*)           carve((size_t)T_TOK * TOPK * 4);
    float*          wts       = (float*)         carve((size_t)T_TOK * TOPK * 4);
    float*          scores    = (float*)         carve((size_t)T_TOK * NE * 4);
    int*            counts    = (int*)           carve(NE * 4);
    int*            offsets   = (int*)           carve((NE + 1) * 4);
    int*            cntShared = (int*)           carve(4);
    int*            offShared = (int*)           carve(4);
    int*            lists     = (int*)           carve((size_t)NE * T_TOK * 4);
    float*          wlists    = (float*)         carve((size_t)NE * T_TOK * 4);
    int*            ident     = (int*)           carve((size_t)T_TOK * 4);
    unsigned short* Hr        = (unsigned short*)carve((size_t)T_TOK * TOPK * HID * 2);
    unsigned short* Hs        = (unsigned short*)carve((size_t)T_TOK * SHID * 2);

    zero_kernel<<<(out_size + 255) / 256, 256, 0, stream>>>(out, out_size);
    gate_kernel<<<T_TOK, 256, 0, stream>>>(x, gw, idx, wts, scores);
    build_lists_kernel<<<1, 32, 0, stream>>>(idx, wts, lists, wlists, counts, offsets,
                                             ident, cntShared, offShared);
    aux_kernel<<<1, 256, 0, stream>>>(scores, counts, out + (size_t)T_TOK * DIM);

    // routed experts: h pass then combine pass
    moe_h_kernel<<<dim3(T_TOK / 128, HID / 64, NE), 256, 0, stream>>>(
        x, w1, w3, lists, wlists, counts, offsets, Hr, HID, (long long)HID * DIM);
    // shared expert as "expert 0" with identity list / unit weights
    moe_h_kernel<<<dim3(T_TOK / 128, SHID / 64, 1), 256, 0, stream>>>(
        x, sw1, sw3, ident, nullptr, cntShared, offShared, Hs, SHID, 0);

    moe_out_kernel<<<dim3(T_TOK / 128, DIM / 64, NE), 256, 0, stream>>>(
        Hr, w2, lists, counts, offsets, out, HID, (long long)DIM * HID);
    moe_out_kernel<<<dim3(T_TOK / 128, DIM / 64, 1), 256, 0, stream>>>(
        Hs, sw2, ident, cntShared, offShared, out, SHID, 0);
}
